// ConvBnReLU3DSparse_52149492908562
// MI455X (gfx1250) — matrix operand order repaired and checked by reading, not compiled, not run on hardware
//
#include <hip/hip_runtime.h>

typedef __attribute__((ext_vector_type(16))) _Float16 v16h;
typedef __attribute__((ext_vector_type(8)))  _Float16 v8h;
typedef __attribute__((ext_vector_type(8)))  float    v8f;

#define CIN   64
#define COUT  64
#define KVOL  27
#define EPSF  1e-5f
#define MTILE 32
#define APAD  72   // 64 f16 payload + 8 f16 pad -> 144B row stride (36 banks, conflict-free)

// ---------------- prep: feats f32 -> f16 with zero row at index 0 ----------------
__global__ void __launch_bounds__(256)
prep_feats(const float* __restrict__ feats, _Float16* __restrict__ featsH, int n) {
  long tid = (long)blockIdx.x * blockDim.x + threadIdx.x;   // each handles 8 elements
  long total = (long)n * CIN / 8;
  if (tid < 8) {                       // zero row 0 (idx -1 sentinel row)
    v8h z = {};
    *(v8h*)(featsH + tid * 8) = z;
  }
  if (tid >= total) return;
  const float* src = feats + tid * 8;
  _Float16* dst = featsH + CIN + tid * 8;   // rows shifted by +1
  v8h out;
#pragma unroll
  for (int i = 0; i < 8; ++i) out[i] = (_Float16)src[i];
  *(v8h*)dst = out;
}

// ------- prep: weight [K][CIN][COUT] f32 -> weightT [K][COUT][CIN] f16 (transposed) -------
__global__ void __launch_bounds__(256)
prep_weight(const float* __restrict__ w, _Float16* __restrict__ wT) {
  int tid = blockIdx.x * blockDim.x + threadIdx.x;
  if (tid >= KVOL * CIN * COUT) return;
  int k   = tid / (CIN * COUT);
  int rem = tid % (CIN * COUT);
  int d   = rem / CIN;   // cout
  int c   = rem % CIN;   // cin
  wT[tid] = (_Float16)w[k * CIN * COUT + c * COUT + d];
}

// ---------------- gather (async->LDS, double buffered) + WMMA sparse conv ----------------
__global__ void __launch_bounds__(256)
conv_wmma(const _Float16* __restrict__ featsH,   // [(N+1)][64] f16, row 0 = zeros
          const _Float16* __restrict__ wT,       // [K][COUT][CIN] f16
          const int* __restrict__ nbr,           // [K][N]
          float* __restrict__ out, int n) {
  __shared__ _Float16 aTile[2][MTILE * APAD];

  const int tid    = threadIdx.x;
  const int lane   = tid & 31;
  const int wave   = tid >> 5;          // 0..7
  const int laneLo = lane & 15;
  const int laneHi = lane >> 4;         // 0/1
  const int mtile  = wave >> 2;         // 0/1 : which 16-row half
  const int ctile  = wave & 3;          // 0..3: which 16-cout tile
  const int rowBase = blockIdx.x * MTILE;

  // gather mapping: 256 threads = 32 rows x 8 x 16B chunks
  const int gr = tid >> 3;              // 0..31 row
  const int gp = tid & 7;               // 0..7  chunk

  // LDS byte offsets for this thread's async destination (addr[31:0] of generic ptr)
  const unsigned ldsOff0 = (unsigned)(size_t)&aTile[0][gr * APAD + gp * 8];
  const unsigned ldsOff1 = (unsigned)(size_t)&aTile[1][gr * APAD + gp * 8];

  // A fragment row (ISA 16-bit A 16x32 layout): lane L holds row M=L%16
  const int aRow = mtile * 16 + laneLo;
  // B fragment column (B 32x16, this code's contiguous k order): lane n holds column n, 16 consecutive K
  const _Float16* wCol = wT + (size_t)(ctile * 16 + laneLo) * CIN;

  const int gRow = rowBase + gr;

  v8f acc = {};

  // --- issue async gather for offset 0 into buffer 0 ---
  {
    int idx = (gRow < n) ? nbr[gRow] : -1;
    const _Float16* src = featsH + (long)(idx + 1) * CIN + gp * 8;
    asm volatile("global_load_async_to_lds_b128 %0, %1, off"
                 :: "v"(ldsOff0), "v"(src) : "memory");
  }

  for (int k = 0; k < KVOL; ++k) {
    const int buf = k & 1;
    // my async store to LDS for buffer `buf` has landed:
    asm volatile("s_wait_asynccnt 0x0" ::: "memory");
    // everyone's stores have landed (and prior consumers of buf^1 are done):
    __syncthreads();

    // prefetch next offset's gather into the other buffer (overlaps with WMMAs)
    if (k + 1 < KVOL) {
      int idx = (gRow < n) ? nbr[(long)(k + 1) * n + gRow] : -1;
      const _Float16* src = featsH + (long)(idx + 1) * CIN + gp * 8;
      asm volatile("global_load_async_to_lds_b128 %0, %1, off"
                   :: "v"(buf ? ldsOff0 : ldsOff1), "v"(src) : "memory");
    }

    const _Float16* aTileB = aTile[buf];
    const _Float16* wBase  = wCol + (size_t)k * CIN * COUT;
#pragma unroll
    for (int kc = 0; kc < 2; ++kc) {
      // A: lane<16 -> K = kc*32 + 0..15; lane>=16 -> kc*32 + 16..31
      union { v16h v; v8h h[2]; } af;
      const _Float16* aBase = aTileB + aRow * APAD + kc * 32 + laneHi * 16;
      af.h[0] = *(const v8h*)(aBase);
      af.h[1] = *(const v8h*)(aBase + 8);
      // B: lane<16 -> K = kc*32 + 0..15; lane>=16 -> kc*32 + 16..31 (contiguous in wT)
      v16h bf = *(const v16h*)(wBase + kc * 32 + laneHi * 16);
      acc = __builtin_amdgcn_wmma_f32_16x16x32_f16(false, af.v, false, bf,
                                                   (short)0, acc, false, false);
    }
  }

  // C/D layout: lane<16 VGPR v -> (M=v, N=lane); lane>=16 -> (M=v+8, N=lane-16)
  const int outCol = ctile * 16 + laneLo;
  const int mBase  = rowBase + mtile * 16 + laneHi * 8;
#pragma unroll
  for (int v = 0; v < 8; ++v) {
    int r = mBase + v;
    if (r < n) out[(long)r * COUT + outCol] = acc[v];
  }
}

// ---------------- BN stats: deterministic two-stage reduction ----------------
__global__ void __launch_bounds__(256)
stats_partial(const float* __restrict__ out, float* __restrict__ partial,
              int n, int rowsPerBlock) {
  __shared__ float sS[256], sQ[256];
  int c = threadIdx.x & 63;
  int g = threadIdx.x >> 6;                 // 0..3
  int base = blockIdx.x * rowsPerBlock;
  int end  = base + rowsPerBlock; if (end > n) end = n;
  float s = 0.f, q = 0.f;
  for (int r = base + g; r < end; r += 4) {
    float x = out[(long)r * COUT + c];
    s += x; q += x * x;
  }
  sS[threadIdx.x] = s; sQ[threadIdx.x] = q;
  __syncthreads();
  if (g == 0) {
    s = sS[c] + sS[c + 64] + sS[c + 128] + sS[c + 192];
    q = sQ[c] + sQ[c + 64] + sQ[c + 128] + sQ[c + 192];
    partial[blockIdx.x * 128 + c]       = s;
    partial[blockIdx.x * 128 + 64 + c]  = q;
  }
}

__global__ void
stats_finalize(const float* __restrict__ partial, const float* __restrict__ gamma,
               const float* __restrict__ beta, float* __restrict__ sb,
               int nBlocks, int n) {
  int c = threadIdx.x;   // 64 threads
  float s = 0.f, q = 0.f;
  for (int b = 0; b < nBlocks; ++b) {
    s += partial[b * 128 + c];
    q += partial[b * 128 + 64 + c];
  }
  float mean  = s / (float)n;
  float var   = q / (float)n - mean * mean;       // population var (ddof=0)
  float scale = gamma[c] * rsqrtf(var + EPSF);
  sb[c]       = scale;
  sb[64 + c]  = beta[c] - mean * scale;
}

// ---------------- fused BN + ReLU, in-place on d_out ----------------
__global__ void __launch_bounds__(256)
bn_relu(float* __restrict__ out, const float* __restrict__ sb, long total4) {
  long tid = (long)blockIdx.x * blockDim.x + threadIdx.x;
  if (tid >= total4) return;
  float4 x = *(float4*)(out + tid * 4);
  int c0 = (int)((tid * 4) & 63);
  x.x = fmaxf(0.f, x.x * sb[c0]     + sb[64 + c0]);
  x.y = fmaxf(0.f, x.y * sb[c0 + 1] + sb[64 + c0 + 1]);
  x.z = fmaxf(0.f, x.z * sb[c0 + 2] + sb[64 + c0 + 2]);
  x.w = fmaxf(0.f, x.w * sb[c0 + 3] + sb[64 + c0 + 3]);
  *(float4*)(out + tid * 4) = x;
}

extern "C" void kernel_launch(void* const* d_in, const int* in_sizes, int n_in,
                              void* d_out, int out_size, void* d_ws, size_t ws_size,
                              hipStream_t stream) {
  const float* feats  = (const float*)d_in[0];
  const float* weight = (const float*)d_in[1];
  const float* gamma  = (const float*)d_in[2];
  const float* beta   = (const float*)d_in[3];
  const int*   nbr    = (const int*)d_in[4];
  float* out = (float*)d_out;
  const int n = in_sizes[0] / CIN;

  // workspace carve-up (256B aligned)
  char* ws = (char*)d_ws;
  size_t featsBytes = (((size_t)(n + 1) * CIN * 2) + 255) & ~(size_t)255;
  size_t wBytes     = (((size_t)KVOL * CIN * COUT * 2) + 255) & ~(size_t)255;
  const int statBlocks = 120;
  size_t pBytes     = (((size_t)statBlocks * 128 * 4) + 255) & ~(size_t)255;
  _Float16* featsH  = (_Float16*)ws;
  _Float16* wT      = (_Float16*)(ws + featsBytes);
  float*    partial = (float*)(ws + featsBytes + wBytes);
  float*    sb      = (float*)(ws + featsBytes + wBytes + pBytes);

  { // feats f32 -> f16 (+zero sentinel row)
    long threads = (long)n * CIN / 8;
    int blocks = (int)((threads + 255) / 256);
    prep_feats<<<blocks, 256, 0, stream>>>(feats, featsH, n);
  }
  { // weight transpose + convert
    int total = KVOL * CIN * COUT;
    prep_weight<<<(total + 255) / 256, 256, 0, stream>>>(weight, wT);
  }
  { // gather + WMMA conv
    int blocks = (n + MTILE - 1) / MTILE;
    conv_wmma<<<blocks, 256, 0, stream>>>(featsH, wT, nbr, out, n);
  }
  { // BN statistics
    int rowsPerBlock = (n + statBlocks - 1) / statBlocks;
    stats_partial<<<statBlocks, 256, 0, stream>>>(out, partial, n, rowsPerBlock);
    stats_finalize<<<1, 64, 0, stream>>>(partial, gamma, beta, sb, statBlocks, n);
  }
  { // fused BN+ReLU in place
    long total4 = (long)n * COUT / 4;
    bn_relu<<<(int)((total4 + 255) / 256), 256, 0, stream>>>(out, sb, total4);
  }
}
